// PocketEncoder_46076409151883
// MI455X (gfx1250) — compile-verified, matrix operand order checked by reading, not run on hardware
//
#include <hip/hip_runtime.h>

#define P_NODES 20000
#define E_EDGES 320000
#define D_DIM   64
#define H_DIM   256
#define OUT_DIM 128
#define LN_EPS  1e-5f

typedef __attribute__((ext_vector_type(16))) __bf16 v16bf;
typedef __attribute__((ext_vector_type(8)))  float  v8f;

union Frag {
  v16bf v;
  unsigned short u[16];
  uint4 q[2];
};

__device__ __forceinline__ unsigned short f2bf(float f) {
  union { float f; unsigned u; } x; x.f = f;
  unsigned r = x.u + 0x7FFFu + ((x.u >> 16) & 1u);   // round-to-nearest-even
  return (unsigned short)(r >> 16);
}

__device__ __forceinline__ float silu_f(float x) { return x / (1.f + __expf(-x)); }
__device__ __forceinline__ float sigm_f(float x) { return 1.f / (1.f + __expf(-x)); }

__device__ __forceinline__ v16bf load_frag(const unsigned short* p) {
  Frag f;
  f.q[0] = ((const uint4*)p)[0];
  f.q[1] = ((const uint4*)p)[1];
  return f.v;
}

__device__ __forceinline__ v8f wmma_bf16(v16bf a, v16bf b, v8f c) {
  // D = A(16x32 bf16) * B(32x16 bf16) + C(16x16 f32)
  return __builtin_amdgcn_wmma_f32_16x16x32_bf16(false, a, false, b, (short)0, c,
                                                 false, false);
}

// reduce across the 16 lanes of a half-wave (rows live per-half, so this
// fully reduces a row's 16 column-partials)
__device__ __forceinline__ float half_reduce(float v) {
  v += __shfl_xor(v, 1, 32);
  v += __shfl_xor(v, 2, 32);
  v += __shfl_xor(v, 4, 32);
  v += __shfl_xor(v, 8, 32);
  return v;
}

// ---------------------------------------------------------------------------
// Pack a row-major f32 [K,N] weight into per-lane WMMA B-fragment order:
// dst[((kt*NT+nt)*32 + L)*16 + j] = bf16( W[kt*32 + (L/16)*16 + j][nt*16 + L%16] )
// ---------------------------------------------------------------------------
__global__ void k_pack(const float* __restrict__ src, unsigned short* __restrict__ dst,
                       int KT, int NT, int N) {
  int tid = blockIdx.x * blockDim.x + threadIdx.x;
  int total = KT * NT * 512;
  if (tid >= total) return;
  int j  = tid & 15;
  int L  = (tid >> 4) & 31;
  int nt = (tid >> 9) % NT;
  int kt = tid / (512 * NT);
  int k = kt * 32 + (L >> 4) * 16 + j;
  int n = nt * 16 + (L & 15);
  dst[tid] = f2bf(src[(size_t)k * N + n]);
}

// h = aa_embed[aa_ids] + ca @ coord_w + coord_b  (f32 + bf16 shadow copy)
__global__ void k_init_h(const int* __restrict__ aa_ids, const float* __restrict__ ca,
                         const float* __restrict__ aa_embed,
                         const float* __restrict__ coord_w, const float* __restrict__ coord_b,
                         float* __restrict__ h, unsigned short* __restrict__ hbf) {
  int tid = blockIdx.x * blockDim.x + threadIdx.x;
  if (tid >= P_NODES * D_DIM) return;
  int p = tid >> 6, d = tid & 63;
  float v = aa_embed[aa_ids[p] * D_DIM + d] + coord_b[d];
  v += ca[p * 3 + 0] * coord_w[0 * D_DIM + d];
  v += ca[p * 3 + 1] * coord_w[1 * D_DIM + d];
  v += ca[p * 3 + 2] * coord_w[2 * D_DIM + d];
  h[tid] = v;
  hbf[tid] = f2bf(v);
}

__global__ void k_zero(float* p, int n) {
  int t = blockIdx.x * blockDim.x + threadIdx.x;
  if (t < n) p[t] = 0.f;
}

__global__ void k_cvt_bf(const float* __restrict__ src, unsigned short* __restrict__ dst, int n) {
  int t = blockIdx.x * blockDim.x + threadIdx.x;
  if (t < n) dst[t] = f2bf(src[t]);
}

// ---------------------------------------------------------------------------
// Fused edge pass: one block = 64 edges (4 WMMA M-tiles), 8 waves, wave w
// owns columns [32w, 32w+32).  B-fragments are loaded once per K-step and
// reused across the 4 M-tiles (4x less L2 weight traffic).  msg2 never
// leaves the accumulators: attention is a register dot + shfl reduction,
// and the segment-sum scatter issues f32 atomics straight from registers.
// ---------------------------------------------------------------------------
__global__ __launch_bounds__(256)
void k_edge(const int* __restrict__ edges, const float* __restrict__ ca,
            const unsigned short* __restrict__ hbf,
            const unsigned short* __restrict__ w1p, const unsigned short* __restrict__ w2p,
            const float* __restrict__ w1raw,   // edge_w1 layer base [193,256]
            const float* __restrict__ b1, const float* __restrict__ b2,
            const float* __restrict__ attw, const float* __restrict__ attb,
            float* __restrict__ agg) {
  __shared__ int   s_nr[64], s_nc[64];
  __shared__ float s_d2[64], s_d1[64];
  __shared__ __align__(16) unsigned short s_m1[64 * H_DIM];  // msg1 bf16, 32 KB
  __shared__ float s_red[8 * 64];                            // per-wave att partials
  __shared__ float s_att[64];

  int tid = threadIdx.x;
  int e0 = blockIdx.x * 64;

  if (tid < 64) {
    int e = e0 + tid;
    int r = edges[e], c = edges[E_EDGES + e];
    s_nr[tid] = r; s_nc[tid] = c;
    float dx = ca[r * 3 + 0] - ca[c * 3 + 0];
    float dy = ca[r * 3 + 1] - ca[c * 3 + 1];
    float dz = ca[r * 3 + 2] - ca[c * 3 + 2];
    float d2 = dx * dx + dy * dy + dz * dz;
    s_d2[tid] = d2;
    s_d1[tid] = sqrtf(d2);
  }
  __syncthreads();

  const int wv = tid >> 5, L = tid & 31;
  const int M = L & 15;
  const int half = L >> 4;
  const int mbase = half * 8;            // C/D: VGPR r holds row M = r + 8*half
  const int nt0 = wv * 2, nt1 = nt0 + 1;
  const int c = L & 15;
  const int n0 = nt0 * 16 + c, n1 = nt1 * 16 + c;

  v8f z = {0, 0, 0, 0, 0, 0, 0, 0};
  v8f acc[4][2];
#pragma unroll
  for (int mt = 0; mt < 4; ++mt) { acc[mt][0] = z; acc[mt][1] = z; }

  // ---- GEMM1: K = 128 (h[row] 0..63, h[col] 64..127) ----
#pragma unroll
  for (int kt = 0; kt < 4; ++kt) {
    v16bf bb0 = load_frag(w1p + ((size_t)(kt * 16 + nt0) * 32 + L) * 16);
    v16bf bb1 = load_frag(w1p + ((size_t)(kt * 16 + nt1) * 32 + L) * 16);
    int koff = (kt & 1) * 32 + half * 16;
#pragma unroll
    for (int mt = 0; mt < 4; ++mt) {
      int m = mt * 16 + M;
      int node = (kt < 2) ? s_nr[m] : s_nc[m];
      v16bf a = load_frag(hbf + (size_t)node * D_DIM + koff);
      acc[mt][0] = wmma_bf16(a, bb0, acc[mt][0]);
      acc[mt][1] = wmma_bf16(a, bb1, acc[mt][1]);
    }
  }
  // epilogue 1: + dist_sq*W1[128] + dist*W1[129] + b1, SiLU, -> LDS bf16
  {
    float wd2_0 = w1raw[128 * H_DIM + n0], wd1_0 = w1raw[129 * H_DIM + n0];
    float wd2_1 = w1raw[128 * H_DIM + n1], wd1_1 = w1raw[129 * H_DIM + n1];
    float bb0 = b1[n0], bb1 = b1[n1];
#pragma unroll
    for (int mt = 0; mt < 4; ++mt) {
#pragma unroll
      for (int r = 0; r < 8; ++r) {
        int m = mt * 16 + mbase + r;
        float v0 = silu_f(acc[mt][0][r] + s_d2[m] * wd2_0 + s_d1[m] * wd1_0 + bb0);
        float v1 = silu_f(acc[mt][1][r] + s_d2[m] * wd2_1 + s_d1[m] * wd1_1 + bb1);
        s_m1[m * H_DIM + n0] = f2bf(v0);
        s_m1[m * H_DIM + n1] = f2bf(v1);
      }
    }
  }
  __syncthreads();

  // ---- GEMM2: K = 256 ----
#pragma unroll
  for (int mt = 0; mt < 4; ++mt) { acc[mt][0] = z; acc[mt][1] = z; }
#pragma unroll
  for (int kt = 0; kt < 8; ++kt) {
    v16bf bb0 = load_frag(w2p + ((size_t)(kt * 16 + nt0) * 32 + L) * 16);
    v16bf bb1 = load_frag(w2p + ((size_t)(kt * 16 + nt1) * 32 + L) * 16);
    int koff = kt * 32 + half * 16;
#pragma unroll
    for (int mt = 0; mt < 4; ++mt) {
      v16bf a = load_frag(s_m1 + (mt * 16 + M) * H_DIM + koff);
      acc[mt][0] = wmma_bf16(a, bb0, acc[mt][0]);
      acc[mt][1] = wmma_bf16(a, bb1, acc[mt][1]);
    }
  }

  // epilogue 2: SiLU in-register; attention partials via shfl reduction
  {
    float bb0 = b2[n0], bb1 = b2[n1];
    float aw0 = attw[n0], aw1 = attw[n1];
#pragma unroll
    for (int mt = 0; mt < 4; ++mt) {
#pragma unroll
      for (int r = 0; r < 8; ++r) {
        float v0 = silu_f(acc[mt][0][r] + bb0);
        float v1 = silu_f(acc[mt][1][r] + bb1);
        acc[mt][0][r] = v0;
        acc[mt][1][r] = v1;
        float p = half_reduce(v0 * aw0 + v1 * aw1);   // row sum over this wave's 32 cols
        if (c == 0) s_red[wv * 64 + mt * 16 + mbase + r] = p;
      }
    }
  }
  __syncthreads();

  if (tid < 64) {
    float s = attb[0];
#pragma unroll
    for (int w = 0; w < 8; ++w) s += s_red[w * 64 + tid];
    s_att[tid] = sigm_f(s);
  }
  __syncthreads();

  // ---- scaled segment-sum scatter straight from accumulators ----
#pragma unroll
  for (int mt = 0; mt < 4; ++mt) {
#pragma unroll
    for (int r = 0; r < 8; ++r) {
      int m = mt * 16 + mbase + r;
      float a = s_att[m];
      float* dst = agg + (size_t)s_nr[m] * H_DIM;
      atomicAdd(dst + n0, acc[mt][0][r] * a);
      atomicAdd(dst + n1, acc[mt][1][r] * a);
    }
  }
}

// ---------------------------------------------------------------------------
// Fused node pass: 32 nodes/block (2 M-tiles). concat(h,agg)[32,320] @ W1 ->
// SiLU -> [32,256] @ W2 -> + residual -> LayerNorm -> h (f32 + bf16) in place.
// ---------------------------------------------------------------------------
__global__ __launch_bounds__(256)
void k_node(const float* __restrict__ h, const unsigned short* __restrict__ hbf,
            const unsigned short* __restrict__ aggbf,
            const unsigned short* __restrict__ nw1p, const unsigned short* __restrict__ nw2p,
            const float* __restrict__ nb1, const float* __restrict__ nb2,
            const float* __restrict__ lng, const float* __restrict__ lnb,
            float* __restrict__ h_out, unsigned short* __restrict__ hbf_out) {
  __shared__ __align__(16) unsigned short s_act[32 * H_DIM];  // 16 KB
  __shared__ __align__(16) float          s_hr[32 * D_DIM];   // 8 KB
  __shared__ float s_mu[32], s_rs[32];

  int tid = threadIdx.x;
  int p0 = blockIdx.x * 32;
  const int wv = tid >> 5, L = tid & 31;
  const int M = L & 15, half = L >> 4, mbase = half * 8;
  const int c = L & 15;
  const int nt0 = wv * 2, nt1 = nt0 + 1;

  v8f z = {0, 0, 0, 0, 0, 0, 0, 0};
  v8f acc[2][2];
  acc[0][0] = z; acc[0][1] = z; acc[1][0] = z; acc[1][1] = z;

  // ---- GEMM1: K = 320 = 64(h) + 256(agg) ----
#pragma unroll
  for (int kt = 0; kt < 10; ++kt) {
    v16bf bb0 = load_frag(nw1p + ((size_t)(kt * 16 + nt0) * 32 + L) * 16);
    v16bf bb1 = load_frag(nw1p + ((size_t)(kt * 16 + nt1) * 32 + L) * 16);
#pragma unroll
    for (int mt = 0; mt < 2; ++mt) {
      int p = p0 + mt * 16 + M;
      v16bf a = (kt < 2)
        ? load_frag(hbf + (size_t)p * D_DIM + kt * 32 + half * 16)
        : load_frag(aggbf + (size_t)p * H_DIM + (kt - 2) * 32 + half * 16);
      acc[mt][0] = wmma_bf16(a, bb0, acc[mt][0]);
      acc[mt][1] = wmma_bf16(a, bb1, acc[mt][1]);
    }
  }
  {
    int n0 = nt0 * 16 + c, n1 = nt1 * 16 + c;
    float bb0 = nb1[n0], bb1 = nb1[n1];
#pragma unroll
    for (int mt = 0; mt < 2; ++mt) {
#pragma unroll
      for (int r = 0; r < 8; ++r) {
        int m = mt * 16 + mbase + r;
        s_act[m * H_DIM + n0] = f2bf(silu_f(acc[mt][0][r] + bb0));
        s_act[m * H_DIM + n1] = f2bf(silu_f(acc[mt][1][r] + bb1));
      }
    }
  }
  __syncthreads();

  // ---- GEMM2: [32,256] @ [256,64]; wave -> (mt = wv/4, nt = wv%4) ----
  {
    int nt = wv & 3, mt = wv >> 2;
    v8f a2 = z;
#pragma unroll
    for (int kt = 0; kt < 8; ++kt) {
      v16bf a = load_frag(s_act + (mt * 16 + M) * H_DIM + kt * 32 + half * 16);
      v16bf b = load_frag(nw2p + ((size_t)(kt * 4 + nt) * 32 + L) * 16);
      a2 = wmma_bf16(a, b, a2);
    }
    int n = nt * 16 + c;
    float bb = nb2[n];
#pragma unroll
    for (int r = 0; r < 8; ++r) {
      int m = mt * 16 + mbase + r;
      s_hr[m * D_DIM + n] = a2[r] + bb + h[(size_t)(p0 + m) * D_DIM + n];
    }
  }
  __syncthreads();

  // ---- LayerNorm stats (one thread per row) ----
  if (tid < 32) {
    float mu = 0.f;
    for (int d = 0; d < D_DIM; ++d) mu += s_hr[tid * D_DIM + d];
    mu *= (1.f / D_DIM);
    float var = 0.f;
    for (int d = 0; d < D_DIM; ++d) { float t = s_hr[tid * D_DIM + d] - mu; var += t * t; }
    var *= (1.f / D_DIM);
    s_mu[tid] = mu;
    s_rs[tid] = rsqrtf(var + LN_EPS);
  }
  __syncthreads();

  for (int q = tid; q < 32 * D_DIM; q += 256) {
    int m = q >> 6, d = q & 63;
    float v = (s_hr[q] - s_mu[m]) * s_rs[m] * lng[d] + lnb[d];
    size_t idx = (size_t)(p0 + m) * D_DIM + d;
    h_out[idx] = v;
    hbf_out[idx] = f2bf(v);
  }
}

// mean over nodes (per-block partial reduce + 64 atomics per block)
__global__ __launch_bounds__(256)
void k_pool_partial(const float* __restrict__ h, float* __restrict__ mean) {
  __shared__ float red[256];
  int tid = threadIdx.x;
  int d = tid & 63;
  int sub = tid >> 6;  // 0..3
  float s = 0.f;
  for (int p = blockIdx.x * 4 + sub; p < P_NODES; p += gridDim.x * 4)
    s += h[(size_t)p * D_DIM + d];
  red[tid] = s;
  __syncthreads();
  if (tid < 64) {
    float t = red[tid] + red[tid + 64] + red[tid + 128] + red[tid + 192];
    atomicAdd(&mean[tid], t);
  }
}

__global__ void k_pool_final(const float* __restrict__ mean, const float* __restrict__ pool_w,
                             const float* __restrict__ pool_b, float* __restrict__ out) {
  int o = blockIdx.x * blockDim.x + threadIdx.x;
  if (o >= OUT_DIM) return;
  float s = pool_b[o];
  for (int d = 0; d < D_DIM; ++d)
    s += (mean[d] * (1.f / P_NODES)) * pool_w[d * OUT_DIM + o];
  out[o] = silu_f(s);
}

// ---------------------------------------------------------------------------
extern "C" void kernel_launch(void* const* d_in, const int* in_sizes, int n_in,
                              void* d_out, int out_size, void* d_ws, size_t ws_size,
                              hipStream_t stream) {
  (void)in_sizes; (void)n_in; (void)out_size; (void)ws_size;
  const int*   aa_ids  = (const int*)d_in[0];
  const float* ca      = (const float*)d_in[1];
  const int*   edges   = (const int*)d_in[2];
  const float* aa_emb  = (const float*)d_in[3];
  const float* coord_w = (const float*)d_in[4];
  const float* coord_b = (const float*)d_in[5];
  const float* edge_w1 = (const float*)d_in[6];
  const float* edge_b1 = (const float*)d_in[7];
  const float* edge_w2 = (const float*)d_in[8];
  const float* edge_b2 = (const float*)d_in[9];
  const float* att_w   = (const float*)d_in[10];
  const float* att_b   = (const float*)d_in[11];
  const float* node_w1 = (const float*)d_in[12];
  const float* node_b1 = (const float*)d_in[13];
  const float* node_w2 = (const float*)d_in[14];
  const float* node_b2 = (const float*)d_in[15];
  const float* ln_g    = (const float*)d_in[16];
  const float* ln_b    = (const float*)d_in[17];
  const float* pool_w  = (const float*)d_in[18];
  const float* pool_b  = (const float*)d_in[19];
  float* out = (float*)d_out;

  char* ws = (char*)d_ws;
  size_t off = 0;
  auto alloc = [&](size_t bytes) -> void* {
    void* p = ws + off;
    off = (off + bytes + 255) & ~(size_t)255;
    return p;
  };
  float*          h     = (float*)alloc((size_t)P_NODES * D_DIM * 4);
  unsigned short* hbf   = (unsigned short*)alloc((size_t)P_NODES * D_DIM * 2);
  float*          agg   = (float*)alloc((size_t)P_NODES * H_DIM * 4);
  unsigned short* aggbf = (unsigned short*)alloc((size_t)P_NODES * H_DIM * 2);
  float*          mean  = (float*)alloc(D_DIM * 4);
  unsigned short* w1p[2], *w2p[2], *nw1p[2], *nw2p[2];
  for (int i = 0; i < 2; ++i) {
    w1p[i]  = (unsigned short*)alloc((size_t)4  * 16 * 512 * 2);  // K=128,N=256
    w2p[i]  = (unsigned short*)alloc((size_t)8  * 16 * 512 * 2);  // K=256,N=256
    nw1p[i] = (unsigned short*)alloc((size_t)10 * 16 * 512 * 2);  // K=320,N=256
    nw2p[i] = (unsigned short*)alloc((size_t)8  * 4  * 512 * 2);  // K=256,N=64
  }

  // pack weights (reads only the non-zero-feature rows of edge_w1)
  for (int i = 0; i < 2; ++i) {
    k_pack<<<(4 * 16 * 512 + 255) / 256, 256, 0, stream>>>(edge_w1 + (size_t)i * 193 * 256, w1p[i], 4, 16, 256);
    k_pack<<<(8 * 16 * 512 + 255) / 256, 256, 0, stream>>>(edge_w2 + (size_t)i * 256 * 256, w2p[i], 8, 16, 256);
    k_pack<<<(10 * 16 * 512 + 255) / 256, 256, 0, stream>>>(node_w1 + (size_t)i * 320 * 256, nw1p[i], 10, 16, 256);
    k_pack<<<(8 * 4 * 512 + 255) / 256, 256, 0, stream>>>(node_w2 + (size_t)i * 256 * 64, nw2p[i], 8, 4, 64);
  }

  k_init_h<<<(P_NODES * D_DIM + 255) / 256, 256, 0, stream>>>(aa_ids, ca, aa_emb, coord_w, coord_b, h, hbf);

  for (int i = 0; i < 2; ++i) {
    k_zero<<<(P_NODES * H_DIM + 255) / 256, 256, 0, stream>>>(agg, P_NODES * H_DIM);
    k_edge<<<E_EDGES / 64, 256, 0, stream>>>(
        edges, ca, hbf, w1p[i], w2p[i],
        edge_w1 + (size_t)i * 193 * 256,
        edge_b1 + i * 256, edge_b2 + i * 256,
        att_w + i * 256, att_b + i, agg);
    k_cvt_bf<<<(P_NODES * H_DIM + 255) / 256, 256, 0, stream>>>(agg, aggbf, P_NODES * H_DIM);
    k_node<<<P_NODES / 32, 256, 0, stream>>>(
        h, hbf, aggbf, nw1p[i], nw2p[i],
        node_b1 + i * 256, node_b2 + i * 64,
        ln_g + i * 64, ln_b + i * 64, h, hbf);
  }

  k_zero<<<1, 64, 0, stream>>>(mean, D_DIM);
  k_pool_partial<<<80, 256, 0, stream>>>(h, mean);
  k_pool_final<<<1, OUT_DIM, 0, stream>>>(mean, pool_w, pool_b, out);
}
